// LSTM_56547539419401
// MI455X (gfx1250) — matrix operand order repaired and checked by reading, not compiled, not run on hardware
//
#include <hip/hip_runtime.h>

#define IN_DIM 256
#define HID    512
#define OUT_DIM 256
#define BATCH  256
#define SEQ    512
#define KTOT   768          // IN_DIM + HID, fused K
#define NKC    (KTOT / 32)  // 24 K-chunks of 32
#define LDS_B_ROWS 64       // 4 gates x 16 hidden rows staged per block

typedef __attribute__((ext_vector_type(16))) __bf16 v16bf;
typedef __attribute__((ext_vector_type(8)))  __bf16 v8bf;
typedef __attribute__((ext_vector_type(8)))  float  v8f;

typedef __attribute__((ext_vector_type(4))) unsigned int v4u;
typedef __attribute__((ext_vector_type(8))) int          v8i;
typedef __attribute__((ext_vector_type(4))) int          v4i;

union AFrag { v16bf v; v8bf h[2]; };

__device__ __forceinline__ unsigned short f2bf(float f) {
  unsigned int u = __float_as_uint(f);
  u += 0x7FFFu + ((u >> 16) & 1u);       // round-to-nearest-even
  return (unsigned short)(u >> 16);
}

__device__ __forceinline__ float sigmoid_f(float x) {
  return 1.0f / (1.0f + __expf(-x));
}

// ---------------------------------------------------------------------------
// f32 -> bf16 bulk conversion (b128 load, b64 store)
// ---------------------------------------------------------------------------
__global__ void cvt_f32_bf16_kernel(const float* __restrict__ src,
                                    unsigned short* __restrict__ dst, int n4) {
  int i = blockIdx.x * blockDim.x + threadIdx.x;
  if (i >= n4) return;
  float4 v = ((const float4*)src)[i];
  ushort4 o;
  o.x = f2bf(v.x); o.y = f2bf(v.y); o.z = f2bf(v.z); o.w = f2bf(v.w);
  ((ushort4*)dst)[i] = o;
}

// Build W_cat[2048][768] bf16: row (q*512+j) = [W_qx[j][0:256] | W_qh[j][0:512]]
// gate order q: g, i, f, o
__global__ void cvt_wcat_kernel(const float* __restrict__ Wgx, const float* __restrict__ Wgh,
                                const float* __restrict__ Wix, const float* __restrict__ Wih,
                                const float* __restrict__ Wfx, const float* __restrict__ Wfh,
                                const float* __restrict__ Wox, const float* __restrict__ Woh,
                                unsigned short* __restrict__ Wcat, int n) {
  int idx = blockIdx.x * blockDim.x + threadIdx.x;
  if (idx >= n) return;
  int row = idx / KTOT;            // 0..2047
  int k   = idx - row * KTOT;      // 0..767
  int q   = row >> 9;              // /512
  int j   = row & 511;
  const float* Wx4[4] = {Wgx, Wix, Wfx, Wox};
  const float* Wh4[4] = {Wgh, Wih, Wfh, Woh};
  float v = (k < IN_DIM) ? Wx4[q][j * IN_DIM + k]
                         : Wh4[q][j * HID + (k - IN_DIM)];
  Wcat[idx] = f2bf(v);
}

__global__ void init_state_kernel(unsigned short* __restrict__ h0,
                                  float* __restrict__ c, int n) {
  int i = blockIdx.x * blockDim.x + threadIdx.x;
  if (i >= n) return;
  h0[i] = 0;
  c[i]  = 0.0f;
}

// ---------------------------------------------------------------------------
// One LSTM timestep:  z = [x_t | h] @ Wcat^T ; gates ; c,h update
//
// Block mapping: jtile = blockIdx & 31 (shared by all 8 waves), btile =
// (blockIdx>>5)*8 + wave.  The 64 weight rows (4 gates x 16 hidden) for the
// block's jtile are staged ONCE into LDS by the Tensor Data Mover (one 2D D#
// per gate), cutting L2 weight traffic 8x; B fragments then come from
// ds_load_b128.  A fragments (x_t / h rows, per-wave private) stay global.
// Grid: 64 blocks x 256 threads = 512 waves = (256/16)*(512/16) tiles.
// ---------------------------------------------------------------------------
__global__ void __launch_bounds__(256, 1)
lstm_step_kernel(const unsigned short* __restrict__ x_bf,   // [B,S,IN] bf16
                 const unsigned short* __restrict__ Wcat,   // [2048,768] bf16
                 const float* __restrict__ bg, const float* __restrict__ bi,
                 const float* __restrict__ bf_, const float* __restrict__ bo,
                 const unsigned short* __restrict__ h_in,   // [B,H] bf16
                 unsigned short* __restrict__ h_out,        // [B,H] bf16
                 float* __restrict__ c,                     // [B,H] f32 (in/out)
                 int s) {
  __shared__ unsigned short Bs[LDS_B_ROWS * KTOT];          // 96 KB

  const int lane   = threadIdx.x & 31;
  const int wave   = threadIdx.x >> 5;
  const int jtile  = blockIdx.x & 31;                // shared across block
  const int btile  = (blockIdx.x >> 5) * 8 + wave;   // 0..15
  const int m_base = btile * 16;
  const int j_base = jtile * 16;

  const int lrow   = lane & 15;
  const int hiHalf = (lane >> 4) & 1;

  // ---- stage B tile (64 rows x 768 bf16) into LDS ----
#if __has_builtin(__builtin_amdgcn_tensor_load_to_lds)
  if (wave == 0) {
    // low 32 bits of a generic LDS pointer == LDS byte offset
    const unsigned ldsBase = (unsigned)(unsigned long long)(void*)&Bs[0];
#pragma unroll
    for (int q = 0; q < 4; ++q) {
      const unsigned long long gaddr =
          (unsigned long long)(const void*)(Wcat + (size_t)(q * HID + j_base) * KTOT);
      // D# group0: count=1 | lds_addr | global_addr[56:0] | type=2
      v4u g0 = { 1u,
                 ldsBase + (unsigned)(q * 16 * KTOT * 2),
                 (unsigned)(gaddr & 0xFFFFFFFFull),
                 (unsigned)(((gaddr >> 32) & 0x01FFFFFFull) | 0x80000000u) };
      // D# group1: data_size=1(2B); tensor_dim0=768; tensor_dim1=16;
      //            tile_dim0=768; tile_dim1=16; tensor_dim0_stride=768
      v8i g1 = { (int)(1u << 16),
                 (int)(768u << 16),
                 (int)(16u << 16),
                 (int)(768u << 16),
                 16,
                 768,
                 0, 0 };
      v4i z4 = {0, 0, 0, 0};
#if defined(__clang_major__) && (__clang_major__ >= 23)
      v8i z8 = {0, 0, 0, 0, 0, 0, 0, 0};
      __builtin_amdgcn_tensor_load_to_lds(g0, g1, z4, z4, z8, 0);
#else
      __builtin_amdgcn_tensor_load_to_lds(g0, g1, z4, z4, 0);
#endif
    }
    __builtin_amdgcn_s_wait_tensorcnt(0);
  }
  __syncthreads();
#else
  // Fallback: cooperative global->LDS copy (uint2 = 4 bf16 per transfer)
  {
    const int total4 = LDS_B_ROWS * KTOT / 4;           // 12288
    for (int i4 = threadIdx.x; i4 < total4; i4 += 256) {
      const int e  = i4 * 4;
      const int rr = e / KTOT;                          // 0..63
      const int kk = e - rr * KTOT;
      const int q  = rr >> 4;
      const int r  = rr & 15;
      const unsigned short* srcp = Wcat + (size_t)(q * HID + j_base + r) * KTOT + kk;
      *(uint2*)&Bs[rr * KTOT + kk] = *(const uint2*)srcp;
    }
  }
  __syncthreads();
#endif

  // A-matrix (16x32 bf16) per-lane: row = lane&15; lanes<16 hold K 0..15,
  // lanes>=16 hold K 16..31 (matches B order)
  const int m = m_base + lrow;
  const unsigned short* xrow = x_bf + ((size_t)m * SEQ + s) * IN_DIM;
  const unsigned short* hrow = h_in + (size_t)m * HID;
  const int aoff = hiHalf ? 16 : 0;

  // B-matrix (32x16 bf16) per-lane from LDS: col = staged row (q*16 + lane&15);
  // lanes<16 hold K 0..15, lanes>=16 hold K 16..31 -> 32B contiguous (2x ds_load_b128)
  const unsigned short* bls[4];
#pragma unroll
  for (int q = 0; q < 4; ++q)
    bls[q] = &Bs[(q * 16 + lrow) * KTOT + (hiHalf ? 16 : 0)];

  // Warm the epilogue's c-state tile while the K loop runs (global_prefetch_b8)
  __builtin_prefetch(c + (size_t)(m_base + hiHalf * 8) * HID + j_base + lrow, 0, 0);

  v8f acc[4] = {};

#pragma unroll
  for (int kc = 0; kc < NKC; ++kc) {
    AFrag a;
    const unsigned short* abase = (kc < IN_DIM / 32)
                                      ? (xrow + kc * 32)
                                      : (hrow + (kc - IN_DIM / 32) * 32);
    a.h[0] = *(const v8bf*)(abase + aoff);
    a.h[1] = *(const v8bf*)(abase + 8 + aoff);
#pragma unroll
    for (int q = 0; q < 4; ++q) {
      v16bf b = *(const v16bf*)(bls[q] + kc * 32);
      acc[q] = __builtin_amdgcn_wmma_f32_16x16x32_bf16(
          false, a.v, false, b, (short)0, acc[q], false, false);
    }
  }

  // Epilogue: gates + state update.
  // C/D layout: VGPR r holds M = r (lanes 0-15) or 8+r (lanes 16-31), N = lane&15.
  const int n = j_base + lrow;
  const float bgv = bg[n], biv = bi[n], bfv = bf_[n], bov = bo[n];
#pragma unroll
  for (int r = 0; r < 8; ++r) {
    const int mr = m_base + hiHalf * 8 + r;
    const float g  = tanhf(acc[0][r] + bgv);
    const float ig = sigmoid_f(acc[1][r] + biv);
    const float fg = sigmoid_f(acc[2][r] + bfv);
    const float og = sigmoid_f(acc[3][r] + bov);
    const size_t idx = (size_t)mr * HID + n;
    const float cn = g * ig + c[idx] * fg;
    c[idx] = cn;
    h_out[idx] = f2bf(tanhf(cn) * og);
  }
}

// ---------------------------------------------------------------------------
// Final projection: out = h_T @ W_ph^T + b_p  (256x256, K=512)
// ---------------------------------------------------------------------------
__global__ void __launch_bounds__(256)
proj_kernel(const unsigned short* __restrict__ h_bf,    // [B,HID] bf16
            const unsigned short* __restrict__ Wph_bf,  // [OUT,HID] bf16
            const float* __restrict__ bp,
            float* __restrict__ out) {                  // [B,OUT] f32
  const int lane   = threadIdx.x & 31;
  const int wave   = threadIdx.x >> 5;
  const int tile   = blockIdx.x * 8 + wave;   // 0..255
  const int btile  = tile >> 4;
  const int otile  = tile & 15;
  const int m_base = btile * 16;
  const int n_base = otile * 16;
  const int lrow   = lane & 15;
  const int hiHalf = (lane >> 4) & 1;

  const unsigned short* arow = h_bf + (size_t)(m_base + lrow) * HID;
  const unsigned short* brow = Wph_bf + (size_t)(n_base + lrow) * HID + (hiHalf ? 16 : 0);
  const int aoff = hiHalf ? 16 : 0;

  v8f acc = {};
  AFrag a;
#pragma unroll
  for (int kc = 0; kc < HID / 32; ++kc) {
    const int k0 = kc * 32;
    a.h[0] = *(const v8bf*)(arow + k0 + aoff);
    a.h[1] = *(const v8bf*)(arow + k0 + 8 + aoff);
    v16bf b = *(const v16bf*)(brow + k0);
    acc = __builtin_amdgcn_wmma_f32_16x16x32_bf16(
        false, a.v, false, b, (short)0, acc, false, false);
  }
  const int n = n_base + lrow;
  const float bpv = bp[n];
#pragma unroll
  for (int r = 0; r < 8; ++r) {
    const int mr = m_base + hiHalf * 8 + r;
    out[(size_t)mr * OUT_DIM + n] = acc[r] + bpv;
  }
}

// ---------------------------------------------------------------------------
extern "C" void kernel_launch(void* const* d_in, const int* in_sizes, int n_in,
                              void* d_out, int out_size, void* d_ws, size_t ws_size,
                              hipStream_t stream) {
  (void)in_sizes; (void)n_in; (void)out_size; (void)ws_size;

  const float* x   = (const float*)d_in[0];
  const float* Wgx = (const float*)d_in[1];
  const float* Wgh = (const float*)d_in[2];
  const float* bg  = (const float*)d_in[3];
  const float* Wix = (const float*)d_in[4];
  const float* Wih = (const float*)d_in[5];
  const float* bi  = (const float*)d_in[6];
  const float* Wfx = (const float*)d_in[7];
  const float* Wfh = (const float*)d_in[8];
  const float* bf_ = (const float*)d_in[9];
  const float* Wox = (const float*)d_in[10];
  const float* Woh = (const float*)d_in[11];
  const float* bo  = (const float*)d_in[12];
  const float* Wph = (const float*)d_in[13];
  const float* bp  = (const float*)d_in[14];
  float* out = (float*)d_out;

  char* ws = (char*)d_ws;
  size_t off = 0;
  auto take = [&](size_t bytes) -> char* {
    char* p = ws + off;
    off = (off + bytes + 255) & ~(size_t)255;
    return p;
  };

  unsigned short* x_bf  = (unsigned short*)take((size_t)BATCH * SEQ * IN_DIM * 2); // 64 MB
  unsigned short* Wcat  = (unsigned short*)take((size_t)4 * HID * KTOT * 2);       //  3 MB
  unsigned short* WphB  = (unsigned short*)take((size_t)OUT_DIM * HID * 2);
  unsigned short* h0    = (unsigned short*)take((size_t)BATCH * HID * 2);
  unsigned short* h1    = (unsigned short*)take((size_t)BATCH * HID * 2);
  float*          cbuf  = (float*)take((size_t)BATCH * HID * 4);

  // --- one-time bf16 conversions ---
  {
    int n4 = BATCH * SEQ * IN_DIM / 4;
    cvt_f32_bf16_kernel<<<(n4 + 255) / 256, 256, 0, stream>>>(x, x_bf, n4);
  }
  {
    int n = 4 * HID * KTOT;
    cvt_wcat_kernel<<<(n + 255) / 256, 256, 0, stream>>>(Wgx, Wgh, Wix, Wih,
                                                         Wfx, Wfh, Wox, Woh, Wcat, n);
  }
  {
    int n4 = OUT_DIM * HID / 4;
    cvt_f32_bf16_kernel<<<(n4 + 255) / 256, 256, 0, stream>>>(Wph, WphB, n4);
  }
  {
    int n = BATCH * HID;
    init_state_kernel<<<(n + 255) / 256, 256, 0, stream>>>(h0, cbuf, n);
  }

  // --- sequential recurrence: 512 fused GEMM+gate steps (h ping-pong) ---
  unsigned short* hin = h0;
  unsigned short* hout = h1;
  for (int s = 0; s < SEQ; ++s) {
    lstm_step_kernel<<<64, 256, 0, stream>>>(x_bf, Wcat, bg, bi, bf_, bo,
                                             hin, hout, cbuf, s);
    unsigned short* t = hin; hin = hout; hout = t;
  }

  // --- final projection (hin now holds h_T) ---
  proj_kernel<<<32, 256, 0, stream>>>(hin, WphB, bp, out);
}
